// SelfAttention_62311385530803
// MI455X (gfx1250) — matrix operand order repaired and checked by reading, not compiled, not run on hardware
//
#include <hip/hip_runtime.h>
#include <hip/hip_bf16.h>

#define B_  8
#define S_  2048
#define DIN 512
#define DH  512
#define M_  (B_*S_)

typedef __bf16 bf16x16 __attribute__((ext_vector_type(16)));
typedef float  f32x8   __attribute__((ext_vector_type(8)));

union Frag {
  bf16x16 v;
  unsigned u32[8];
};

__device__ inline unsigned short bf16bits(float f) {
  __bf16 h = (__bf16)f;
  return __builtin_bit_cast(unsigned short, h);
}

__device__ inline unsigned pack2(float lo, float hi) {
  return (unsigned)bf16bits(lo) | ((unsigned)bf16bits(hi) << 16);
}

// ---------------------------------------------------------------------------
// Kernel 1: QKV projection GEMM.  C[M,N] = x[M,K] @ W[K,N] + b, M=16384,
// N=K=512.  64x64 tile per workgroup, 8 waves, each wave does two 16x16
// WMMA accumulators.  Output bf16 into workspace.
// ---------------------------------------------------------------------------
__global__ __launch_bounds__(256) void qkv_gemm(
    const float* __restrict__ x,
    const float* __restrict__ Wq, const float* __restrict__ bq,
    const float* __restrict__ Wk, const float* __restrict__ bk,
    const float* __restrict__ Wv, const float* __restrict__ bv,
    unsigned short* __restrict__ Qo, unsigned short* __restrict__ Ko,
    unsigned short* __restrict__ Vo)
{
  const float* W; const float* bias; unsigned short* out;
  if (blockIdx.z == 0)      { W = Wq; bias = bq; out = Qo; }
  else if (blockIdx.z == 1) { W = Wk; bias = bk; out = Ko; }
  else                      { W = Wv; bias = bv; out = Vo; }

  __shared__ unsigned short lA[64 * 32];   // [row][k] bf16
  __shared__ unsigned short lB[64 * 32];   // transposed: [n][k] bf16

  const int tid  = threadIdx.x;
  const int lane = tid & 31;
  const int w    = tid >> 5;
  const int wm   = w & 3;            // m-subtile 0..3
  const int wn0  = (w >> 2) * 2;     // base n-subtile: 0 or 2
  const int tileM = blockIdx.y * 64;
  const int tileN = blockIdx.x * 64;

  f32x8 acc0 = {}, acc1 = {};

  for (int kk = 0; kk < DIN / 32; ++kk) {
    const int k0 = kk * 32;
    // stage A: 64x32 f32 -> bf16
    #pragma unroll
    for (int i = 0; i < 8; ++i) {
      int idx = tid + i * 256;
      int r = idx >> 5, c = idx & 31;
      lA[idx] = bf16bits(x[(size_t)(tileM + r) * DIN + k0 + c]);
    }
    // stage B: 32x64 f32, stored transposed [n][k]
    #pragma unroll
    for (int i = 0; i < 8; ++i) {
      int idx = tid + i * 256;
      int k = idx >> 6, n = idx & 63;
      lB[n * 32 + k] = bf16bits(W[(size_t)(k0 + k) * DH + tileN + n]);
    }
    __syncthreads();

    // A fragment: 16x32, row = wm*16 + lane%16
    Frag a;
    {
      int row  = wm * 16 + (lane & 15);
      int koff = (lane >> 4) * 8;
      #pragma unroll
      for (int j = 0; j < 8; ++j) {
        int k = (j < 4 ? 2 * j : 16 + 2 * (j - 4)) + koff;
        a.u32[j] = *(const unsigned*)&lA[row * 32 + k];
      }
    }
    // B fragments + WMMA
    {
      int khalf = (lane >> 4) * 8;
      int n0 = (wn0 + 0) * 16 + (lane & 15);
      int n1 = (wn0 + 1) * 16 + (lane & 15);
      Frag b0, b1;
      #pragma unroll
      for (int j = 0; j < 8; ++j) {
        int k = khalf + (j < 4 ? 2 * j : 16 + 2 * (j - 4));
        b0.u32[j] = *(const unsigned*)&lB[n0 * 32 + k];
        b1.u32[j] = *(const unsigned*)&lB[n1 * 32 + k];
      }
      acc0 = __builtin_amdgcn_wmma_f32_16x16x32_bf16(false, a.v, false, b0.v,
                                                     (short)0, acc0, false, false);
      acc1 = __builtin_amdgcn_wmma_f32_16x16x32_bf16(false, a.v, false, b1.v,
                                                     (short)0, acc1, false, false);
    }
    __syncthreads();
  }

  // epilogue: bias, convert, store bf16
  {
    int rbase = tileM + wm * 16 + ((lane >> 4) * 8);
    int c0 = tileN + (wn0 + 0) * 16 + (lane & 15);
    int c1 = tileN + (wn0 + 1) * 16 + (lane & 15);
    float bb0 = bias[c0], bb1 = bias[c1];
    #pragma unroll
    for (int r = 0; r < 8; ++r) {
      int row = rbase + r;
      out[(size_t)row * DH + c0] = bf16bits(acc0[r] + bb0);
      out[(size_t)row * DH + c1] = bf16bits(acc1[r] + bb1);
    }
  }
}

// ---------------------------------------------------------------------------
// Kernel 2: flash attention.  WG = 256 threads = 8 waves = 2 q-groups of
// 16 rows x 4 waves.  Each wave holds its 16x512 Q slab resident as WMMA
// fragments and owns a 128-wide D_H chunk of the context accumulator.
// K tile is staged raw into LDS with CDNA5 async-to-LDS copies (ASYNCcnt);
// V tile is staged transposed for contiguous B-fragment reads.
// ---------------------------------------------------------------------------
__global__ __launch_bounds__(256, 1) void flash_attn(
    const unsigned short* __restrict__ Q,
    const unsigned short* __restrict__ K,
    const unsigned short* __restrict__ V,
    float* __restrict__ out)
{
  __shared__ unsigned short klds[64 * 512];   // K tile raw: [key][dh]
  __shared__ unsigned short vlds[64 * 512];   // V tile transposed: [dh][key]
  __shared__ float slds[2][16 * 64];          // scores per q-group
  __shared__ float stat_m[2][16];
  __shared__ float stat_a[2][16];
  __shared__ float stat_l[2][16];

  const int tid   = threadIdx.x;
  const int lane  = tid & 31;
  const int w     = tid >> 5;
  const int g     = w >> 2;      // q-group 0/1
  const int wsub  = w & 3;       // key subtile / dh chunk
  const int batch = blockIdx.y;
  const int qbase = blockIdx.x * 32 + g * 16;

  const unsigned short* Qb = Q + (size_t)batch * S_ * DH;
  const unsigned short* Kb = K + (size_t)batch * S_ * DH;
  const unsigned short* Vb = V + (size_t)batch * S_ * DH;

  const float scale = 0.044194173824159216f;   // 1/sqrt(512)

  // resident Q fragments: 16 dh-steps of 32
  Frag qf[16];
  {
    int row  = qbase + (lane & 15);
    int koff = (lane >> 4) * 8;
    #pragma unroll
    for (int f = 0; f < 16; ++f) {
      #pragma unroll
      for (int j = 0; j < 8; ++j) {
        int k = f * 32 + (j < 4 ? 2 * j : 16 + 2 * (j - 4)) + koff;
        qf[f].u32[j] = *(const unsigned*)(Qb + (size_t)row * DH + k);
      }
    }
  }

  f32x8 cacc[8];
  #pragma unroll
  for (int t = 0; t < 8; ++t) cacc[t] = (f32x8){};

  float m_run = -1e30f, l_run = 0.0f;

  for (int kt = 0; kt < S_ / 64; ++kt) {
    const int kbase = kt * 64;

    // ---- stage K tile (raw, contiguous 64KB) via async copy to LDS ------
    {
      const unsigned short* gbase = Kb + (size_t)kbase * DH;
      #pragma unroll 4
      for (int i = 0; i < 16; ++i) {
        int idx = tid + i * 256;                  // b128 chunk index, 4096 total
        unsigned ldsa = (unsigned)(uintptr_t)(klds + idx * 8);
        unsigned voff = (unsigned)idx * 16u;      // byte offset from gbase
        asm volatile("global_load_async_to_lds_b128 %0, %1, %2"
                     :: "v"(ldsa), "v"(voff), "s"(gbase) : "memory");
      }
    }

    // ---- stage V tile 64x512 transposed into LDS -------------------------
    for (int i = 0; i < 64; ++i) {
      int idx = tid + i * 256;        // uint-pair index, 16384 total
      int key = idx >> 8;
      int dp  = idx & 255;
      unsigned pv = *(const unsigned*)(Vb + (size_t)(kbase + key) * DH + dp * 2);
      vlds[(dp * 2 + 0) * 64 + key] = (unsigned short)(pv & 0xFFFFu);
      vlds[(dp * 2 + 1) * 64 + key] = (unsigned short)(pv >> 16);
    }

    // prefetch next K/V tiles into cache while we compute on this one
    if (kt + 1 < S_ / 64) {
      const unsigned short* nk = Kb + (size_t)(kbase + 64) * DH;
      const unsigned short* nv = Vb + (size_t)(kbase + 64) * DH;
      __builtin_prefetch(nk + (size_t)tid * 128, 0, 0);   // 256B per thread
      __builtin_prefetch(nv + (size_t)tid * 128, 0, 0);
    }

    // wait for this wave's async K copies, then make staging WG-visible
    asm volatile("s_wait_asynccnt 0x0" ::: "memory");
    __syncthreads();

    // ---- score subtile: this wave covers keys [kbase + wsub*16, +16) ----
    {
      f32x8 sacc = {};
      int keyl  = wsub * 16 + (lane & 15);      // key within tile
      int khalf = (lane >> 4) * 8;
      #pragma unroll
      for (int f = 0; f < 16; ++f) {
        Frag bf;
        #pragma unroll
        for (int j = 0; j < 8; ++j)
          bf.u32[j] = *(const unsigned*)&klds[keyl * DH + f * 32 + khalf + (j < 4 ? 2 * j : 16 + 2 * (j - 4))];
        sacc = __builtin_amdgcn_wmma_f32_16x16x32_bf16(false, qf[f].v, false, bf.v,
                                                       (short)0, sacc, false, false);
      }
      int rb  = (lane >> 4) * 8;
      int col = wsub * 16 + (lane & 15);
      #pragma unroll
      for (int r = 0; r < 8; ++r)
        slds[g][(rb + r) * 64 + col] = sacc[r] * scale;
    }
    __syncthreads();

    // ---- online-softmax stats (one wave per q-group, lanes 0..15) -------
    if (wsub == 0 && lane < 16) {
      int row = lane;
      float tmax = -1e30f;
      for (int c = 0; c < 64; ++c) tmax = fmaxf(tmax, slds[g][row * 64 + c]);
      float mnew  = fmaxf(m_run, tmax);
      float alpha = __expf(m_run - mnew);
      float lsum  = 0.f;
      for (int c = 0; c < 64; ++c) lsum += __expf(slds[g][row * 64 + c] - mnew);
      l_run = l_run * alpha + lsum;
      m_run = mnew;
      stat_m[g][row] = mnew;
      stat_a[g][row] = alpha;
    }
    __syncthreads();

    // ---- rescale context accumulator by alpha[row] ----------------------
    {
      int rb = (lane >> 4) * 8;
      float al[8];
      #pragma unroll
      for (int r = 0; r < 8; ++r) al[r] = stat_a[g][rb + r];
      #pragma unroll
      for (int t = 0; t < 8; ++t)
        #pragma unroll
        for (int r = 0; r < 8; ++r)
          cacc[t][r] *= al[r];
    }

    // ---- build P fragments (16x64 -> two 16x32 bf16 A-fragments) --------
    Frag pf[2];
    {
      int row  = lane & 15;
      int koff = (lane >> 4) * 8;
      float mrow = stat_m[g][row];
      #pragma unroll
      for (int h = 0; h < 2; ++h) {
        #pragma unroll
        for (int j = 0; j < 8; ++j) {
          int k = h * 32 + (j < 4 ? 2 * j : 16 + 2 * (j - 4)) + koff;
          float p0 = __expf(slds[g][row * 64 + k]     - mrow);
          float p1 = __expf(slds[g][row * 64 + k + 1] - mrow);
          pf[h].u32[j] = pack2(p0, p1);
        }
      }
    }

    // ---- context: P(16x64) @ V(64x128 chunk) ----------------------------
    {
      int khalf = (lane >> 4) * 8;
      #pragma unroll
      for (int n = 0; n < 8; ++n) {
        int dh = wsub * 128 + n * 16 + (lane & 15);
        #pragma unroll
        for (int h = 0; h < 2; ++h) {
          Frag vf;
          #pragma unroll
          for (int j = 0; j < 8; ++j)
            vf.u32[j] = *(const unsigned*)&vlds[dh * 64 + h * 32 + khalf + (j < 4 ? 2 * j : 16 + 2 * (j - 4))];
          cacc[n] = __builtin_amdgcn_wmma_f32_16x16x32_bf16(false, pf[h].v, false, vf.v,
                                                            (short)0, cacc[n], false, false);
        }
      }
    }
    __syncthreads();
  }

  if (wsub == 0 && lane < 16) stat_l[g][lane] = l_run;
  __syncthreads();

  // ---- epilogue: divide by l, store fp32 --------------------------------
  {
    int rb = (lane >> 4) * 8;
    float inv[8];
    #pragma unroll
    for (int r = 0; r < 8; ++r) inv[r] = 1.0f / stat_l[g][rb + r];
    #pragma unroll
    for (int t = 0; t < 8; ++t) {
      int dh = wsub * 128 + t * 16 + (lane & 15);
      #pragma unroll
      for (int r = 0; r < 8; ++r) {
        int row = qbase + rb + r;
        out[((size_t)batch * S_ + row) * DH + dh] = cacc[t][r] * inv[r];
      }
    }
  }
}

extern "C" void kernel_launch(void* const* d_in, const int* in_sizes, int n_in,
                              void* d_out, int out_size, void* d_ws, size_t ws_size,
                              hipStream_t stream)
{
  (void)in_sizes; (void)n_in; (void)out_size; (void)ws_size;
  const float* x  = (const float*)d_in[0];
  const float* Wq = (const float*)d_in[1];
  const float* bq = (const float*)d_in[2];
  const float* Wk = (const float*)d_in[3];
  const float* bk = (const float*)d_in[4];
  const float* Wv = (const float*)d_in[5];
  const float* bv = (const float*)d_in[6];
  float* out = (float*)d_out;

  unsigned short* qws = (unsigned short*)d_ws;
  unsigned short* kws = qws + (size_t)M_ * DH;
  unsigned short* vws = kws + (size_t)M_ * DH;

  qkv_gemm<<<dim3(DH / 64, M_ / 64, 3), 256, 0, stream>>>(
      x, Wq, bq, Wk, bk, Wv, bv, qws, kws, vws);
  flash_attn<<<dim3(S_ / 32, B_), 256, 0, stream>>>(qws, kws, vws, out);
}
